// CosineDistance_17660905521623
// MI455X (gfx1250) — matrix operand order repaired and checked by reading, not compiled, not run on hardware
//
#include <hip/hip_runtime.h>
#include <hip/hip_bf16.h>

// ---------------------------------------------------------------------------
// CosineDistance / MaxSim on MI455X (gfx1250, wave32, WMMA)
//
//   out[a,b] = sum_i max_j dot(pano_n[a,i,:], sat_n[b,j,:])
//   a=64, b=256, i=j=64, d=128,  output [64,256] f32
//
// v3: same as v2 (LDS-staged sat_b tile, batched B fragments, 16 WMMA/wave)
// but with the async-to-LDS builtin called with its actual pointer types:
// (addrspace(1) v4i*, addrspace(3) v4i*, imm, imm).
// ---------------------------------------------------------------------------

typedef __attribute__((ext_vector_type(16))) _Float16 v16h;
typedef __attribute__((ext_vector_type(8)))  _Float16 v8h;
typedef __attribute__((ext_vector_type(4)))  _Float16 v4h;
typedef __attribute__((ext_vector_type(8)))  float    v8f;
typedef __attribute__((ext_vector_type(4)))  int      v4i;

#define DDIM 128
#define NTOK 64      // tokens per image (i and j)
#define NA   64      // pano images
#define NB   256     // sat images
#define LDSROW 136   // 128 halves + 8 pad -> 68 dwords/row, bank-conflict-free

#if __has_builtin(__builtin_amdgcn_global_load_async_to_lds_b128)
#define USE_ASYNC_LDS 1
#else
#define USE_ASYNC_LDS 0
#endif

// ---------------------------------------------------------------------------
// Pass 1: L2-normalize each 128-d row and convert to f16.
// One wave32 per row; 4 rows per 128-thread block.
// ---------------------------------------------------------------------------
__global__ __launch_bounds__(128) void norm_rows_f16(
    const float* __restrict__ in, _Float16* __restrict__ out) {
  const int wave = threadIdx.x >> 5;
  const int lane = threadIdx.x & 31;
  const long row = (long)blockIdx.x * 4 + wave;

  const float4 v = *(const float4*)(in + row * DDIM + lane * 4);
  float ss = v.x * v.x + v.y * v.y + v.z * v.z + v.w * v.w;
#pragma unroll
  for (int off = 1; off < 32; off <<= 1)
    ss += __shfl_xor(ss, off, 32);

  // F.normalize semantics: x / max(||x||, eps)
  const float scale = 1.0f / fmaxf(sqrtf(ss), 1e-12f);

  v4h o;
  o[0] = (_Float16)(v.x * scale);
  o[1] = (_Float16)(v.y * scale);
  o[2] = (_Float16)(v.z * scale);
  o[3] = (_Float16)(v.w * scale);
  *(v4h*)(out + row * DDIM + lane * 4) = o;
}

// ---------------------------------------------------------------------------
// Main kernel: one (b,a) pair per block (grid 256 x 64), 4 waves.
// sat_b tile lives in LDS (padded rows); wave w owns i-tile w and computes
// all 4 j-tiles: 16 v_wmma_f32_16x16x32_f16 per wave.
// ---------------------------------------------------------------------------
__global__ __launch_bounds__(128) void maxsim_wmma(
    const _Float16* __restrict__ pano,  // [64][64][128] normalized f16
    const _Float16* __restrict__ sat,   // [256][64][128] normalized f16
    float* __restrict__ out) {          // [64][256]
  __shared__ __align__(16) _Float16 sB[NTOK * LDSROW];

  const int b    = blockIdx.x;
  const int a    = blockIdx.y;
  const int wave = threadIdx.x >> 5;    // i-tile index 0..3
  const int lane = threadIdx.x & 31;
  const int l16  = lane & 15;
  const int hi   = lane >> 4;           // 0 = lanes 0-15, 1 = lanes 16-31

  const _Float16* __restrict__ P = pano + (long)a * NTOK * DDIM;
  const _Float16* __restrict__ S = sat  + (long)b * NTOK * DDIM;

  // ---- stage S_b (64 x 128 f16) into LDS, rows padded to LDSROW halves ----
  // 1024 chunks of 16B; 128 threads x 8 iterations.
#pragma unroll
  for (int it = 0; it < 8; ++it) {
    const int idx = threadIdx.x + it * 128;
    const int row = idx >> 4;             // 16 chunks of 16B per 256B row
    const int col = (idx & 15) * 8;       // half-element offset
#if USE_ASYNC_LDS
    __builtin_amdgcn_global_load_async_to_lds_b128(
        (__attribute__((address_space(1))) v4i*)(S + (long)row * DDIM + col),
        (__attribute__((address_space(3))) v4i*)&sB[row * LDSROW + col],
        0, 0);
#else
    *(float4*)&sB[row * LDSROW + col] =
        *(const float4*)(S + (long)row * DDIM + col);
#endif
  }
#if USE_ASYNC_LDS
#if __has_builtin(__builtin_amdgcn_s_wait_asynccnt)
  __builtin_amdgcn_s_wait_asynccnt(0);
#else
  asm volatile("s_wait_asynccnt 0x0" ::: "memory");
#endif
#endif
  __syncthreads();

  v8f acc[4] = {};   // j-tiles 0..3, each 16x16 f32 tile

  // A fragment (16x32 f16, M x K): lane = row i (mod 16); lanes 0-15 hold
  // K = {0..7, 16..23}, lanes 16-31 hold K = {8..15, 24..31}  (ISA 7.12.2).
  const _Float16* arow = P + (long)(wave * 16 + l16) * DDIM + hi * 8;

#pragma unroll
  for (int ks = 0; ks < 4; ++ks) {
    const int k0 = ks * 32;
    v8h  alo = *(const v8h*)(arow + k0);        // K = k0 + hi*8 + 0..7
    v8h  ahi = *(const v8h*)(arow + k0 + 16);   // K = k0 + 16 + hi*8 + 0..7
    v16h afrag = __builtin_shufflevector(alo, ahi,
        0, 1, 2, 3, 4, 5, 6, 7, 8, 9, 10, 11, 12, 13, 14, 15);

    // Batch all four B fragments (from LDS) before the four WMMAs.
    // B fragment (32x16 f16, K x N): lane = column N = sat token j (mod 16);
    // lanes 0-15 hold K={0..7,16..23}, lanes 16-31 hold K={8..15,24..31}
    // (same K order as A), from LDS row j at offset k0 + hi*8 and +16.
    v16h bf[4];
#pragma unroll
    for (int jt = 0; jt < 4; ++jt) {
      const _Float16* bp = &sB[(jt * 16 + l16) * LDSROW + k0 + hi * 8];
      v8h blo = *(const v8h*)(bp);
      v8h bhi = *(const v8h*)(bp + 16);
      bf[jt] = __builtin_shufflevector(blo, bhi,
          0, 1, 2, 3, 4, 5, 6, 7, 8, 9, 10, 11, 12, 13, 14, 15);
    }
#pragma unroll
    for (int jt = 0; jt < 4; ++jt)
      acc[jt] = __builtin_amdgcn_wmma_f32_16x16x32_f16(
          /*neg_a=*/false, afrag, /*neg_b=*/false, bf[jt],
          /*c_mod=*/(short)0, acc[jt], /*reuse_a=*/false, /*reuse_b=*/false);
  }

  // C/D layout: VGPR r -> i = 16*wave + r + 8*hi (fixed per lane-half),
  //             lane%16 -> j (mod 16), j-tile index = jt.
  // max over j = elementwise max over jt, then xor-butterfly over lanes
  // 1/2/4/8 (stays within each 16-lane half). Then sum over the 8 rows this
  // lane-half owns, then add across halves (xor 16).
  float s = 0.0f;
#pragma unroll
  for (int r = 0; r < 8; ++r) {
    float m = fmaxf(fmaxf(acc[0][r], acc[1][r]), fmaxf(acc[2][r], acc[3][r]));
#pragma unroll
    for (int off = 1; off < 16; off <<= 1)
      m = fmaxf(m, __shfl_xor(m, off, 32));
    s += m;   // row-max for i = 16*wave + r + 8*hi (replicated in half)
  }
  s += __shfl_xor(s, 16, 32);   // sum over this wave's 16 i-rows

  __shared__ float red[4];
  if (lane == 0) red[wave] = s;
  __syncthreads();
  if (threadIdx.x == 0)
    out[(long)a * NB + b] = red[0] + red[1] + red[2] + red[3];
}

// ---------------------------------------------------------------------------
// Launcher. Inputs (setup_inputs order):
//   d_in[0] = sat  f32 [256,64,128], d_in[1] = pano f32 [64,64,128]
// Workspace: pano_f16 (1 MB) then sat_f16 (4 MB).
// ---------------------------------------------------------------------------
extern "C" void kernel_launch(void* const* d_in, const int* in_sizes, int n_in,
                              void* d_out, int out_size, void* d_ws, size_t ws_size,
                              hipStream_t stream) {
  (void)in_sizes; (void)n_in; (void)out_size; (void)ws_size;
  const float* sat_f32  = (const float*)d_in[0];
  const float* pano_f32 = (const float*)d_in[1];
  float* out = (float*)d_out;

  _Float16* pano16 = (_Float16*)d_ws;
  _Float16* sat16  = (_Float16*)((char*)d_ws +
                      (size_t)NA * NTOK * DDIM * sizeof(_Float16));

  norm_rows_f16<<<(NA * NTOK) / 4, 128, 0, stream>>>(pano_f32, pano16);
  norm_rows_f16<<<(NB * NTOK) / 4, 128, 0, stream>>>(sat_f32, sat16);

  dim3 grid(NB, NA);
  maxsim_wmma<<<grid, 128, 0, stream>>>(pano16, sat16, out);
}
